// LinearAttention_5025111736442
// MI455X (gfx1250) — compile-verified, matrix operand order checked by reading, not run on hardware
//
#include <hip/hip_runtime.h>
#include <cstdint>
#include <cstddef>

// ---------------------------------------------------------------------------
// Types for CDNA5 WMMA (gfx1250, wave32)
// ---------------------------------------------------------------------------
typedef __bf16 bf16_t;
typedef __attribute__((ext_vector_type(16))) __bf16 bf16x16;
typedef __attribute__((ext_vector_type(8)))  __bf16 bf16x8;
typedef __attribute__((ext_vector_type(8)))  float  f32x8;

#define DEV __device__ __forceinline__

// D(16x16,f32) = A(16x32,bf16) x B(32x16,bf16) + C
DEV f32x8 wmma_bf16(bf16x16 a, bf16x16 b, f32x8 c) {
    return __builtin_amdgcn_wmma_f32_16x16x32_bf16(
        /*neg_a=*/false, a, /*neg_b=*/false, b,
        /*c_mod=*/(short)0, c, /*reuse_a=*/false, /*reuse_b=*/false);
}

// Per-lane A/B fragment load. `row` points at the 32-element K-block for this
// lane's matrix row (A: row m, row-major; B: column n, i.e. transposed tile).
// half = lane>>4.  Layout per ISA: elems 0..7 <- K=half*8+0..7,
// elems 8..15 <- K=16+half*8+0..7.  Both loads are 16B-aligned b128.
DEV bf16x16 load_frag(const bf16_t* row, int half) {
    bf16x8 lo = *(const bf16x8*)(row + half * 8);
    bf16x8 hi = *(const bf16x8*)(row + 16 + half * 8);
    bf16x16 r;
#pragma unroll
    for (int i = 0; i < 8; ++i) { r[i] = lo[i]; r[i + 8] = hi[i]; }
    return r;
}

DEV f32x8 zero8() { f32x8 z = {0.f,0.f,0.f,0.f,0.f,0.f,0.f,0.f}; return z; }

// Problem constants
#define BATCH 16
#define DIM   256
#define HEADS 4
#define DH    32
#define HID   128
#define NPIX  4096
#define PTILE 64
#define QKSCALE 0.17677669529663687f   // 32^-0.5
#define RMS_MUL 16.0f                  // sqrt(256)
#define EPS_RMS 1e-12f

// ---------------------------------------------------------------------------
// K0: f32 -> bf16 weight conversion
// ---------------------------------------------------------------------------
__global__ __launch_bounds__(256) void cvt_bf16_kernel(const float* __restrict__ s,
                                                       bf16_t* __restrict__ d, int n) {
    int i = blockIdx.x * 256 + threadIdx.x;
    if (i < n) d[i] = (bf16_t)s[i];
}

// ---------------------------------------------------------------------------
// K1: RMSNorm(x,g1) + QKV projection for one (batch, 64-pixel tile).
//   A = w_qkv_bf16 (384x256, row-major, from L2), B = xn tile (256x64) in LDS
//   stored transposed [pix][c] (stride 264 -> 16B aligned frags).
//   Outputs: q raw f32 [b][h][n][d], k raw f32 [b][h][d][n], v bf16 [b][h][e][n]
// ---------------------------------------------------------------------------
#define XN_STRIDE 264
__global__ __launch_bounds__(256) void qkv_kernel(
    const float* __restrict__ x, const float* __restrict__ g1,
    const bf16_t* __restrict__ wq,
    float* __restrict__ q_raw, float* __restrict__ k_raw, bf16_t* __restrict__ v_bf)
{
    __shared__ bf16_t xn[PTILE * XN_STRIDE];
    __shared__ float  g1s[DIM];
    __shared__ float  part[256];
    __shared__ float  pscale[PTILE];

    const int t    = threadIdx.x;
    const int lane = t & 31, wv = t >> 5;
    const int half = (lane >> 4), lm = lane & 15;
    const int b    = blockIdx.y;
    const int p0   = blockIdx.x * PTILE;
    const int grp  = t >> 6, p = t & 63;

    g1s[t] = g1[t];

    // Phase 1: load x tile, per-pixel sum of squares, stash raw bf16 in LDS
    const size_t xbase = (size_t)b * DIM * NPIX + p0;
    float ss = 0.f;
#pragma unroll 8
    for (int cc = 0; cc < 64; ++cc) {
        int c = grp * 64 + cc;
        float v = x[xbase + (size_t)c * NPIX + p];
        ss += v * v;
        xn[p * XN_STRIDE + c] = (bf16_t)v;
    }
    part[t] = ss;
    __syncthreads();
    if (t < PTILE) {
        float tot = part[t] + part[64 + t] + part[128 + t] + part[192 + t];
        pscale[t] = RMS_MUL / fmaxf(sqrtf(tot), EPS_RMS);
    }
    __syncthreads();

    // Phase 2: normalize in place (f32 math, bf16 storage)
    {
        float sc = pscale[p];
#pragma unroll 8
        for (int cc = 0; cc < 64; ++cc) {
            int c = grp * 64 + cc;
            float v = (float)xn[p * XN_STRIDE + c];
            xn[p * XN_STRIDE + c] = (bf16_t)(v * sc * g1s[c]);
        }
    }
    __syncthreads();

    // Phase 3: 384x64 = (24 Mtiles x 4 Ntiles) GEMM, 8 waves x 12 C-tiles
    f32x8 acc[3][4];
#pragma unroll
    for (int mi = 0; mi < 3; ++mi)
#pragma unroll
        for (int ni = 0; ni < 4; ++ni) acc[mi][ni] = zero8();

    for (int ks = 0; ks < 8; ++ks) {            // K = 256, 8 steps of 32
        bf16x16 af[3], bfv[4];
#pragma unroll
        for (int mi = 0; mi < 3; ++mi) {
            int row = (wv * 3 + mi) * 16 + lm;
            af[mi] = load_frag(wq + row * DIM + ks * 32, half);
        }
#pragma unroll
        for (int ni = 0; ni < 4; ++ni) {
            int n = ni * 16 + lm;
            bfv[ni] = load_frag(&xn[n * XN_STRIDE + ks * 32], half);
        }
#pragma unroll
        for (int mi = 0; mi < 3; ++mi)
#pragma unroll
            for (int ni = 0; ni < 4; ++ni)
                acc[mi][ni] = wmma_bf16(af[mi], bfv[ni], acc[mi][ni]);
    }

    // Phase 4: scatter q/k/v
#pragma unroll
    for (int mi = 0; mi < 3; ++mi) {
        int row8 = (wv * 3 + mi) * 16 + half * 8;   // 8 consecutive out rows
#pragma unroll
        for (int ni = 0; ni < 4; ++ni) {
            int n = p0 + ni * 16 + lm;
            f32x8 a = acc[mi][ni];
            if (row8 < HID) {                        // q : [b][h][n][d] raw f32
                int hh = row8 >> 5, d0 = row8 & 31;
                float* dst = q_raw + (((size_t)(b * HEADS + hh) * NPIX) + n) * DH + d0;
                ((float4*)dst)[0] = make_float4(a[0], a[1], a[2], a[3]);
                ((float4*)dst)[1] = make_float4(a[4], a[5], a[6], a[7]);
            } else if (row8 < 2 * HID) {             // k : [b][h][d][n] raw f32
                int kr = row8 - HID, hh = kr >> 5, d0 = kr & 31;
#pragma unroll
                for (int j = 0; j < 8; ++j)
                    k_raw[((size_t)(b * HEADS + hh) * DH + d0 + j) * NPIX + n] = a[j];
            } else {                                 // v : [b][h][e][n] bf16
                int vr = row8 - 2 * HID, hh = vr >> 5, e0 = vr & 31;
#pragma unroll
                for (int j = 0; j < 8; ++j)
                    v_bf[((size_t)(b * HEADS + hh) * DH + e0 + j) * NPIX + n] = (bf16_t)a[j];
            }
        }
    }
}

// ---------------------------------------------------------------------------
// K2: k softmax stats per channel (max, sumexp over 4096 pixels)
// ---------------------------------------------------------------------------
__global__ __launch_bounds__(256) void kstats_kernel(
    const float* __restrict__ k_raw, float* __restrict__ kmax, float* __restrict__ ksum)
{
    __shared__ float red[256];
    const int ch = blockIdx.x, b = blockIdx.y, t = threadIdx.x;
    const float* row = k_raw + (size_t)(b * HID + ch) * NPIX;

    float m = -3.0e38f;
    for (int i = t; i < NPIX; i += 256) m = fmaxf(m, row[i]);
    red[t] = m; __syncthreads();
    for (int s = 128; s > 0; s >>= 1) { if (t < s) red[t] = fmaxf(red[t], red[t + s]); __syncthreads(); }
    float mx = red[0]; __syncthreads();

    float s = 0.f;
    for (int i = t; i < NPIX; i += 256) s += __expf(row[i] - mx);
    red[t] = s; __syncthreads();
    for (int st = 128; st > 0; st >>= 1) { if (t < st) red[t] += red[t + st]; __syncthreads(); }
    if (t == 0) { kmax[b * HID + ch] = mx; ksum[b * HID + ch] = red[0]; }
}

// ---------------------------------------------------------------------------
// K3: context[d][e] = sum_n softmax(k)[d][n] * v[e][n]   (per b, head)
//   M=N=32, K=4096 split over 8 waves; exp computed on the fly in f32,
//   fed to WMMA as bf16. Cross-wave reduce in LDS; store ctx^T bf16 [e][d].
// ---------------------------------------------------------------------------
__global__ __launch_bounds__(256) void ctx_kernel(
    const float* __restrict__ k_raw, const bf16_t* __restrict__ v_bf,
    const float* __restrict__ kmax, const float* __restrict__ ksum,
    bf16_t* __restrict__ ctx_t)
{
    __shared__ float partl[8 * 4 * 256];     // 32 KB
    const int t = threadIdx.x, lane = t & 31, wv = t >> 5;
    const int half = lane >> 4, lm = lane & 15;
    const int hh = blockIdx.x, b = blockIdx.y;
    const size_t base = (size_t)(b * HEADS + hh) * DH;

    float mx[2], rs[2];
#pragma unroll
    for (int dt = 0; dt < 2; ++dt) {
        int d = dt * 16 + lm;
        mx[dt] = kmax[(b * HEADS + hh) * DH + d];
        rs[dt] = 1.0f / ksum[(b * HEADS + hh) * DH + d];
    }

    f32x8 acc[4];
#pragma unroll
    for (int i = 0; i < 4; ++i) acc[i] = zero8();

    for (int kk = 0; kk < 16; ++kk) {        // this wave's K slice: 512
        int kb = wv * 512 + kk * 32;
        bf16x16 af[2], bfv[2];
#pragma unroll
        for (int dt = 0; dt < 2; ++dt) {
            const float* kp = k_raw + (base + dt * 16 + lm) * NPIX + kb;
            bf16x16 f;
#pragma unroll
            for (int i = 0; i < 8; ++i) {
                f[i]     = (bf16_t)(__expf(kp[half * 8 + i]      - mx[dt]) * rs[dt]);
                f[i + 8] = (bf16_t)(__expf(kp[16 + half * 8 + i] - mx[dt]) * rs[dt]);
            }
            af[dt] = f;
        }
#pragma unroll
        for (int et = 0; et < 2; ++et)
            bfv[et] = load_frag(v_bf + (base + et * 16 + lm) * NPIX + kb, half);
#pragma unroll
        for (int dt = 0; dt < 2; ++dt)
#pragma unroll
            for (int et = 0; et < 2; ++et)
                acc[dt * 2 + et] = wmma_bf16(af[dt], bfv[et], acc[dt * 2 + et]);
    }

#pragma unroll
    for (int tile = 0; tile < 4; ++tile)
#pragma unroll
        for (int j = 0; j < 8; ++j)
            partl[(wv * 4 + tile) * 256 + lane * 8 + j] = acc[tile][j];
    __syncthreads();

#pragma unroll
    for (int ii = 0; ii < 4; ++ii) {
        int o = t * 4 + ii;
        int tile = o >> 8, li = o & 255, ln = li >> 3, j = li & 7;
        int dt = tile >> 1, et = tile & 1;
        int d = dt * 16 + (ln >> 4) * 8 + j;
        int e = et * 16 + (ln & 15);
        float s = 0.f;
#pragma unroll
        for (int w = 0; w < 8; ++w) s += partl[(w * 4 + tile) * 256 + li];
        ctx_t[(base + e) * DH + d] = (bf16_t)s;      // ctx^T: [e][d]
    }
}

// ---------------------------------------------------------------------------
// K4: q softmax (local over d) -> hidden = ctx^T @ q -> out = w_out@hidden + b
//     -> RMSNorm(g2) -> final store.  One (batch, 64-pixel tile) per WG.
// ---------------------------------------------------------------------------
#define QL_STRIDE 40     // 80 B rows: 16B aligned, conflict-light
#define OH_STRIDE 136    // 272 B rows: 16B aligned
__global__ __launch_bounds__(256) void out_kernel(
    const float* __restrict__ q_raw, const bf16_t* __restrict__ ctx_t,
    const bf16_t* __restrict__ wo, const float* __restrict__ b_out,
    const float* __restrict__ g2, float* __restrict__ out)
{
    __shared__ bf16_t qls[HEADS * PTILE * QL_STRIDE];   // 20.0 KB
    __shared__ bf16_t oh[PTILE * OH_STRIDE];            // 17.0 KB
    __shared__ float  sq[PTILE];
    __shared__ float  psc[PTILE];

    const int t = threadIdx.x, lane = t & 31, wv = t >> 5;
    const int half = lane >> 4, lm = lane & 15;
    const int b = blockIdx.y, p0 = blockIdx.x * PTILE;

    if (t < PTILE) sq[t] = 0.f;

    // Stage 1: q softmax over d (32) per (head, pixel); one thread each.
    {
        int hh = t >> 6, p = t & 63;
        const float* qp = q_raw + ((size_t)(b * HEADS + hh) * NPIX + p0 + p) * DH;
        float v[DH];
        float m = -3.0e38f;
#pragma unroll
        for (int d = 0; d < DH; ++d) { v[d] = qp[d]; m = fmaxf(m, v[d]); }
        float s = 0.f;
#pragma unroll
        for (int d = 0; d < DH; ++d) { v[d] = __expf(v[d] - m); s += v[d]; }
        float r = QKSCALE / s;
        bf16_t* dst = &qls[(hh * PTILE + p) * QL_STRIDE];
#pragma unroll
        for (int d = 0; d < DH; ++d) dst[d] = (bf16_t)(v[d] * r);
    }
    __syncthreads();

    // Stage 2: hidden[e][n] = sum_d ctx^T[e][d] * q_soft[d][n]  (K=32, 1 step)
    {
        int hh = wv >> 1, mt = wv & 1;
        bf16x16 afr = load_frag(ctx_t + ((size_t)(b * HEADS + hh) * DH + mt * 16 + lm) * DH, half);
#pragma unroll
        for (int nt = 0; nt < 4; ++nt) {
            bf16x16 bfr = load_frag(&qls[(hh * PTILE + nt * 16 + lm) * QL_STRIDE], half);
            f32x8 a = wmma_bf16(afr, bfr, zero8());
            int e_base = hh * DH + mt * 16 + half * 8;
            int n = nt * 16 + lm;
#pragma unroll
            for (int j = 0; j < 8; ++j) oh[n * OH_STRIDE + e_base + j] = (bf16_t)a[j];
        }
    }
    __syncthreads();

    // Stage 3: out[o][n] = w_out[o][c] * hidden[c][n]  (M=256,K=128,N=64)
    f32x8 acc2[2][4];
#pragma unroll
    for (int mi = 0; mi < 2; ++mi)
#pragma unroll
        for (int nt = 0; nt < 4; ++nt) acc2[mi][nt] = zero8();

    for (int ks = 0; ks < 4; ++ks) {
        bf16x16 af[2], bfv[4];
#pragma unroll
        for (int mi = 0; mi < 2; ++mi) {
            int row = wv * 32 + mi * 16 + lm;
            af[mi] = load_frag(wo + row * HID + ks * 32, half);
        }
#pragma unroll
        for (int nt = 0; nt < 4; ++nt)
            bfv[nt] = load_frag(&oh[(nt * 16 + lm) * OH_STRIDE + ks * 32], half);
#pragma unroll
        for (int mi = 0; mi < 2; ++mi)
#pragma unroll
            for (int nt = 0; nt < 4; ++nt)
                acc2[mi][nt] = wmma_bf16(af[mi], bfv[nt], acc2[mi][nt]);
    }

    // Stage 4: + bias, per-pixel sumsq via LDS f32 atomics
    float bb[2][8], gg[2][8];
#pragma unroll
    for (int mi = 0; mi < 2; ++mi) {
        int row8 = wv * 32 + mi * 16 + half * 8;
        float4 b0 = *(const float4*)(b_out + row8);
        float4 b1 = *(const float4*)(b_out + row8 + 4);
        float4 g0 = *(const float4*)(g2 + row8);
        float4 g1v = *(const float4*)(g2 + row8 + 4);
        bb[mi][0] = b0.x; bb[mi][1] = b0.y; bb[mi][2] = b0.z; bb[mi][3] = b0.w;
        bb[mi][4] = b1.x; bb[mi][5] = b1.y; bb[mi][6] = b1.z; bb[mi][7] = b1.w;
        gg[mi][0] = g0.x; gg[mi][1] = g0.y; gg[mi][2] = g0.z; gg[mi][3] = g0.w;
        gg[mi][4] = g1v.x; gg[mi][5] = g1v.y; gg[mi][6] = g1v.z; gg[mi][7] = g1v.w;
    }
#pragma unroll
    for (int nt = 0; nt < 4; ++nt) {
        float s = 0.f;
#pragma unroll
        for (int mi = 0; mi < 2; ++mi) {
            f32x8 a = acc2[mi][nt];
#pragma unroll
            for (int j = 0; j < 8; ++j) { a[j] += bb[mi][j]; s += a[j] * a[j]; }
            acc2[mi][nt] = a;
        }
        atomicAdd(&sq[nt * 16 + lm], s);     // ds_add_f32
    }
    __syncthreads();
    if (t < PTILE) psc[t] = RMS_MUL / fmaxf(sqrtf(sq[t]), EPS_RMS);
    __syncthreads();

    // Stage 5: normalized store (16-lane contiguous rows)
#pragma unroll
    for (int mi = 0; mi < 2; ++mi) {
        int row8 = wv * 32 + mi * 16 + half * 8;
#pragma unroll
        for (int nt = 0; nt < 4; ++nt) {
            int nl = nt * 16 + lm;
            float sc = psc[nl];
            f32x8 a = acc2[mi][nt];
#pragma unroll
            for (int j = 0; j < 8; ++j)
                out[((size_t)b * DIM + row8 + j) * NPIX + p0 + nl] = a[j] * sc * gg[mi][j];
        }
    }
}

// ---------------------------------------------------------------------------
// Host launcher.  Workspace use: ~85 MB (q/k raw f32, v/ctx/weights bf16).
// ---------------------------------------------------------------------------
extern "C" void kernel_launch(void* const* d_in, const int* in_sizes, int n_in,
                              void* d_out, int out_size, void* d_ws, size_t ws_size,
                              hipStream_t stream) {
    (void)in_sizes; (void)n_in; (void)out_size; (void)ws_size;
    const float* x     = (const float*)d_in[0];
    const float* g1    = (const float*)d_in[1];
    const float* w_qkv = (const float*)d_in[2];
    const float* w_out = (const float*)d_in[3];
    const float* b_out = (const float*)d_in[4];
    const float* g2    = (const float*)d_in[5];
    float* out = (float*)d_out;

    char* ws = (char*)d_ws;
    size_t off = 0;
    auto take = [&](size_t bytes) -> char* {
        char* p = ws + off;
        off = (off + bytes + 255) & ~(size_t)255;
        return p;
    };
    bf16_t* wq_bf = (bf16_t*)take((size_t)384 * 256 * sizeof(bf16_t));
    bf16_t* wo_bf = (bf16_t*)take((size_t)256 * 128 * sizeof(bf16_t));
    float*  q_raw = (float*) take((size_t)BATCH * HEADS * NPIX * DH * sizeof(float));
    float*  k_raw = (float*) take((size_t)BATCH * HEADS * DH * NPIX * sizeof(float));
    bf16_t* v_bf  = (bf16_t*)take((size_t)BATCH * HEADS * DH * NPIX * sizeof(bf16_t));
    float*  kmax  = (float*) take((size_t)BATCH * HID * sizeof(float));
    float*  ksum  = (float*) take((size_t)BATCH * HID * sizeof(float));
    bf16_t* ctx_t = (bf16_t*)take((size_t)BATCH * HEADS * DH * DH * sizeof(bf16_t));

    cvt_bf16_kernel<<<dim3(384), 256, 0, stream>>>(w_qkv, wq_bf, 384 * 256);
    cvt_bf16_kernel<<<dim3(128), 256, 0, stream>>>(w_out, wo_bf, 256 * 128);
    qkv_kernel    <<<dim3(64, 16), 256, 0, stream>>>(x, g1, wq_bf, q_raw, k_raw, v_bf);
    kstats_kernel <<<dim3(128, 16), 256, 0, stream>>>(k_raw, kmax, ksum);
    ctx_kernel    <<<dim3(4, 16), 256, 0, stream>>>(k_raw, v_bf, kmax, ksum, ctx_t);
    out_kernel    <<<dim3(64, 16), 256, 0, stream>>>(q_raw, ctx_t, wo_bf, b_out, g2, out);
}
